// GroupedQueryAttention_14594298871883
// MI455X (gfx1250) — matrix operand order repaired and checked by reading, not compiled, not run on hardware
//
#include <hip/hip_runtime.h>
#include <hip/hip_bf16.h>

// Problem constants (from reference)
#define B_   2
#define S_   2048
#define D_   4096
#define H_   32
#define HKV_ 8
#define HD_  128
#define G_   (H_ / HKV_)

typedef __attribute__((ext_vector_type(16))) __bf16 v16bf;
typedef __attribute__((ext_vector_type(8)))  __bf16 v8bf;
typedef __attribute__((ext_vector_type(8)))  float  v8f;

// ---------------------------------------------------------------------------
// WMMA helpers (CDNA5 wave32, V_WMMA_F32_16X16X32_BF16)
// A fragment (16x32 MxK): lane l -> row l&15, khalf = l>>4;
//   elems 0..7 = K[khalf*8 .. +7], elems 8..15 = K[16+khalf*8 .. +7].
// B fragment mirrors A with N in place of M; the right-hand operand is always
// stored transposed (N x K, K contiguous) so one loader serves both.
// ---------------------------------------------------------------------------
static __device__ __forceinline__ v16bf load_frag(const __bf16* tile, int ld) {
  const int lane = threadIdx.x & 31;
  const int row  = lane & 15;
  const int kh   = (lane >> 4) & 1;
  const __bf16* p = tile + (size_t)row * (size_t)ld + kh * 8;
  v8bf lo = *(const v8bf*)(p);
  v8bf hi = *(const v8bf*)(p + 16);
  v16bf r;
#pragma unroll
  for (int i = 0; i < 8; ++i) { r[i] = lo[i]; r[i + 8] = hi[i]; }
  return r;
}

static __device__ __forceinline__ v8f wmma_bf16(v16bf a, v16bf b, v8f c) {
  return __builtin_amdgcn_wmma_f32_16x16x32_bf16(false, a, false, b, (short)0, c,
                                                 false, false);
}

// ---------------------------------------------------------------------------
// Elementwise conversion / transpose kernels
// ---------------------------------------------------------------------------
__global__ void f32_to_bf16(const float* __restrict__ in, __bf16* __restrict__ out,
                            size_t n) {
  size_t i = (size_t)blockIdx.x * blockDim.x + threadIdx.x;
  if (i < n) out[i] = (__bf16)in[i];
}

// W (K x N, row-major f32) -> Wt (N x K, row-major bf16)
__global__ void transpose_f32_to_bf16(const float* __restrict__ W,
                                      __bf16* __restrict__ Wt, int K, int N) {
  size_t i = (size_t)blockIdx.x * blockDim.x + threadIdx.x;
  size_t total = (size_t)K * N;
  if (i >= total) return;
  int k = (int)(i / N);
  int n = (int)(i % N);
  Wt[(size_t)n * K + k] = (__bf16)W[i];
}

// V (B,S,HKV,HD) bf16 -> Vt (B,HKV,HD,S) bf16  (keys contiguous)
__global__ void transpose_v(const __bf16* __restrict__ Vb, __bf16* __restrict__ Vt) {
  size_t i = (size_t)blockIdx.x * blockDim.x + threadIdx.x;
  size_t total = (size_t)B_ * HKV_ * HD_ * S_;
  if (i >= total) return;
  int s = (int)(i % S_);   size_t t = i / S_;
  int d = (int)(t % HD_);  t /= HD_;
  int kv = (int)(t % HKV_);
  int b  = (int)(t / HKV_);
  Vt[i] = Vb[(((size_t)b * S_ + s) * HKV_ + kv) * HD_ + d];
}

// Rotate-half RoPE in place on (B*S, nh, HD) bf16; one thread per (tok,head,j<64)
__global__ void rope_bf16(__bf16* __restrict__ buf, int nh) {
  size_t i = (size_t)blockIdx.x * blockDim.x + threadIdx.x;
  size_t total = (size_t)B_ * S_ * nh * (HD_ / 2);
  if (i >= total) return;
  int j = (int)(i & 63);          // HD/2 = 64
  size_t t = i >> 6;
  int head = (int)(t % nh);
  size_t tok = t / nh;
  int pos = (int)(tok % S_);
  __bf16* p = buf + (tok * nh + head) * HD_;
  float inv = __expf(-(float)j * 0.14391156946f);  // ln(10000)/64
  float ang = (float)pos * inv;
  float c = __cosf(ang), sn = __sinf(ang);
  float t1 = (float)p[j], t2 = (float)p[j + 64];
  p[j]      = (__bf16)(t1 * c - t2 * sn);
  p[j + 64] = (__bf16)(t2 * c + t1 * sn);
}

// ---------------------------------------------------------------------------
// GEMM: C(MxN) = A(MxK) * Bt(NxK)^T   (bf16 inputs, OutT output)
// One wave per block; wave computes a 64x64 tile:
//   4 A-frags x 4 B-frags -> 16 WMMA per 16 b128 loads per k-step of 32.
// ---------------------------------------------------------------------------
template <typename OutT>
__global__ void __launch_bounds__(32)
gemm_bf16_wmma(const __bf16* __restrict__ A, const __bf16* __restrict__ Bt,
               OutT* __restrict__ C, int M, int N, int K) {
  const int lane = threadIdx.x;
  const int nt = blockIdx.x;   // 64-wide column tile
  const int mt = blockIdx.y;   // 64-row tile
  const __bf16* arow = A + (size_t)mt * 64 * K;
  const __bf16* brow = Bt + (size_t)nt * 64 * K;
  v8f acc[4][4] = {};
  for (int k = 0; k < K; k += 32) {
    __builtin_prefetch(arow + k + 32, 0, 0);
    __builtin_prefetch(brow + k + 32, 0, 0);
    v16bf af[4];
#pragma unroll
    for (int i = 0; i < 4; ++i)
      af[i] = load_frag(arow + (size_t)i * 16 * K + k, K);
#pragma unroll
    for (int j = 0; j < 4; ++j) {
      v16bf bf = load_frag(brow + (size_t)j * 16 * K + k, K);
#pragma unroll
      for (int i = 0; i < 4; ++i) acc[i][j] = wmma_bf16(af[i], bf, acc[i][j]);
    }
  }
  const int r0  = ((lane >> 4) & 1) * 8;
  const int col = lane & 15;
#pragma unroll
  for (int i = 0; i < 4; ++i)
#pragma unroll
    for (int j = 0; j < 4; ++j)
#pragma unroll
      for (int r = 0; r < 8; ++r) {
        size_t row = (size_t)mt * 64 + i * 16 + r0 + r;
        size_t cc  = (size_t)nt * 64 + j * 16 + col;
        C[row * N + cc] = (OutT)acc[i][j][r];
      }
}

// ---------------------------------------------------------------------------
// Flash-style causal GQA attention, 32 queries x HD=128 per wave.
// Q tile lives in LDS (pre-scaled by 1/sqrt(HD)) to keep the wave under
// 256 VGPRs (avoids s_set_vgpr_msb churn, raises occupancy); every K-frag
// and V-frag load feeds two query row-tiles (16 WMMA per 8 frag loads).
// ---------------------------------------------------------------------------
__global__ void __launch_bounds__(32)
attn_fwd(const __bf16* __restrict__ Qb, const __bf16* __restrict__ Kb,
         const __bf16* __restrict__ Vt, __bf16* __restrict__ Ao) {
  __shared__ __align__(16) __bf16 qbuf[32 * 128];  // 8 KB: 32 queries x HD
  __shared__ __align__(16) __bf16 pbuf[32 * 32];   // 2 KB: P staging
  const int lane = threadIdx.x;
  int id = blockIdx.x;
  const int qt = id & (S_ / 32 - 1); id >>= 6;   // S/32 = 64
  const int h  = id & (H_ - 1);
  const int b  = id >> 5;
  const int kv = h / G_;
  const int q0 = qt * 32;
  const int rowoff = ((lane >> 4) & 1) * 8;
  const int col = lane & 15;
  const float scale = 0.08838834764831845f;  // 1/sqrt(128)

  const __bf16* qptr  = Qb + (((size_t)(b * S_ + q0)) * H_ + h) * HD_;
  const __bf16* khead = Kb + ((size_t)b * S_ * HKV_ + kv) * HD_;
  const __bf16* vhead = Vt + ((size_t)(b * HKV_ + kv)) * HD_ * S_;

  // Copy this wave's 32x128 Q tile to LDS, folding in the softmax scale.
  {
    const __bf16* src = qptr + (size_t)lane * (H_ * HD_);  // row = lane
#pragma unroll
    for (int c = 0; c < 16; ++c) {
      v8bf v = *(const v8bf*)(src + c * 8);
      v8bf w;
#pragma unroll
      for (int e = 0; e < 8; ++e) w[e] = (__bf16)((float)v[e] * scale);
      *(v8bf*)&qbuf[lane * 128 + c * 8] = w;
    }
  }
  __syncthreads();

  v8f acc[2][8] = {};
  float m[2][8], ssum[2][8];
#pragma unroll
  for (int u = 0; u < 2; ++u)
#pragma unroll
    for (int r = 0; r < 8; ++r) { m[u][r] = -1e30f; ssum[u][r] = 0.0f; }

  for (int kb = 0; kb <= q0 + 31; kb += 32) {
    // ---- scores: 2 q-tiles x 2 key sub-tiles over 32 keys -------------
    v8f s[2][2] = {};
    const __bf16* kp0 = khead + (size_t)kb * (HKV_ * HD_);
    const __bf16* kp1 = kp0 + (size_t)16 * (HKV_ * HD_);
#pragma unroll
    for (int t = 0; t < 4; ++t) {
      v16bf kf0 = load_frag(kp0 + t * 32, HKV_ * HD_);
      v16bf kf1 = load_frag(kp1 + t * 32, HKV_ * HD_);
      v16bf q0f = load_frag(&qbuf[t * 32], 128);
      v16bf q1f = load_frag(&qbuf[16 * 128 + t * 32], 128);
      s[0][0] = wmma_bf16(q0f, kf0, s[0][0]);
      s[0][1] = wmma_bf16(q0f, kf1, s[0][1]);
      s[1][0] = wmma_bf16(q1f, kf0, s[1][0]);
      s[1][1] = wmma_bf16(q1f, kf1, s[1][1]);
    }
    const int key0 = kb + col, key1 = kb + 16 + col;
#pragma unroll
    for (int u = 0; u < 2; ++u) {
      float c0[8], c1[8], mn[8], f[8];
#pragma unroll
      for (int r = 0; r < 8; ++r) {
        const int q = q0 + u * 16 + rowoff + r;
        float v0 = s[u][0][r]; if (key0 > q) v0 = -1e30f;
        float v1 = s[u][1][r]; if (key1 > q) v1 = -1e30f;
        c0[r] = v0; c1[r] = v1;
        float cm = fmaxf(v0, v1);
        cm = fmaxf(cm, __shfl_xor(cm, 1, 32));
        cm = fmaxf(cm, __shfl_xor(cm, 2, 32));
        cm = fmaxf(cm, __shfl_xor(cm, 4, 32));
        cm = fmaxf(cm, __shfl_xor(cm, 8, 32));
        mn[r] = fmaxf(m[u][r], cm);
        f[r]  = __expf(m[u][r] - mn[r]);
        m[u][r] = mn[r];
      }
#pragma unroll
      for (int r = 0; r < 8; ++r) {
        float p0 = __expf(c0[r] - mn[r]);
        float p1 = __expf(c1[r] - mn[r]);
        float t = p0 + p1;
        t += __shfl_xor(t, 1, 32);
        t += __shfl_xor(t, 2, 32);
        t += __shfl_xor(t, 4, 32);
        t += __shfl_xor(t, 8, 32);
        ssum[u][r] = ssum[u][r] * f[r] + t;
        pbuf[(u * 16 + rowoff + r) * 32 + col]      = (__bf16)p0;
        pbuf[(u * 16 + rowoff + r) * 32 + 16 + col] = (__bf16)p1;
      }
#pragma unroll
      for (int j = 0; j < 8; ++j)
#pragma unroll
        for (int r = 0; r < 8; ++r) acc[u][j][r] = acc[u][j][r] * f[r];
    }
    __syncthreads();
    // ---- PV: P tiles (16x32) x Vt slabs (32 keys x 128 d) -------------
    v16bf af0 = load_frag(&pbuf[0], 32);
    v16bf af1 = load_frag(&pbuf[16 * 32], 32);
#pragma unroll
    for (int j = 0; j < 8; ++j) {
      v16bf bv = load_frag(vhead + (size_t)(j * 16) * S_ + kb, S_);
      acc[0][j] = wmma_bf16(af0, bv, acc[0][j]);
      acc[1][j] = wmma_bf16(af1, bv, acc[1][j]);
    }
    __syncthreads();
  }

#pragma unroll
  for (int u = 0; u < 2; ++u) {
    __bf16* optr = Ao + (((size_t)(b * S_ + q0 + u * 16)) * H_ + h) * HD_;
#pragma unroll
    for (int j = 0; j < 8; ++j)
#pragma unroll
      for (int r = 0; r < 8; ++r) {
        float v = acc[u][j][r] / ssum[u][r];
        optr[(size_t)(rowoff + r) * (H_ * HD_) + j * 16 + col] = (__bf16)v;
      }
  }
}

// ---------------------------------------------------------------------------
// Host-side orchestration
// ---------------------------------------------------------------------------
extern "C" void kernel_launch(void* const* d_in, const int* in_sizes, int n_in,
                              void* d_out, int out_size, void* d_ws, size_t ws_size,
                              hipStream_t stream) {
  (void)in_sizes; (void)n_in; (void)out_size; (void)ws_size;
  const float* x  = (const float*)d_in[0];
  const float* Wq = (const float*)d_in[1];
  const float* Wk = (const float*)d_in[2];
  const float* Wv = (const float*)d_in[3];
  const float* Wo = (const float*)d_in[4];
  float* out = (float*)d_out;

  // Workspace layout (bf16 elements). Total ~88.1M elems = ~176 MB.
  const size_t nMD  = (size_t)B_ * S_ * D_;        // x / attn-out (H*HD == D)
  const size_t nWq  = (size_t)D_ * H_ * HD_;
  const size_t nWkv = (size_t)D_ * HKV_ * HD_;
  const size_t nQ   = (size_t)B_ * S_ * H_ * HD_;
  const size_t nKV  = (size_t)B_ * S_ * HKV_ * HD_;

  __bf16* ws  = (__bf16*)d_ws;
  __bf16* xb  = ws;                 // consumed by projections; reused as attn out
  __bf16* WqT = xb  + nMD;
  __bf16* WkT = WqT + nWq;
  __bf16* WvT = WkT + nWkv;
  __bf16* WoT = WvT + nWkv;
  __bf16* Qb  = WoT + nWq;
  __bf16* Kb  = Qb  + nQ;
  __bf16* Vb  = Kb  + nKV;
  __bf16* VT  = Vb  + nKV;
  __bf16* attn = xb;                // alias: x no longer needed by then

  const int TPB = 256;
  f32_to_bf16<<<(unsigned)((nMD + TPB - 1) / TPB), TPB, 0, stream>>>(x, xb, nMD);
  transpose_f32_to_bf16<<<(unsigned)((nWq + TPB - 1) / TPB), TPB, 0, stream>>>(
      Wq, WqT, D_, H_ * HD_);
  transpose_f32_to_bf16<<<(unsigned)((nWkv + TPB - 1) / TPB), TPB, 0, stream>>>(
      Wk, WkT, D_, HKV_ * HD_);
  transpose_f32_to_bf16<<<(unsigned)((nWkv + TPB - 1) / TPB), TPB, 0, stream>>>(
      Wv, WvT, D_, HKV_ * HD_);
  transpose_f32_to_bf16<<<(unsigned)((nWq + TPB - 1) / TPB), TPB, 0, stream>>>(
      Wo, WoT, H_ * HD_, D_);

  // Projections: Q (4096x4096x4096), K/V (4096x1024x4096); 64x64 tiles/wave
  dim3 gq(H_ * HD_ / 64, B_ * S_ / 64);
  gemm_bf16_wmma<__bf16><<<gq, 32, 0, stream>>>(xb, WqT, Qb, B_ * S_, H_ * HD_, D_);
  dim3 gkv(HKV_ * HD_ / 64, B_ * S_ / 64);
  gemm_bf16_wmma<__bf16><<<gkv, 32, 0, stream>>>(xb, WkT, Kb, B_ * S_, HKV_ * HD_, D_);
  gemm_bf16_wmma<__bf16><<<gkv, 32, 0, stream>>>(xb, WvT, Vb, B_ * S_, HKV_ * HD_, D_);

  rope_bf16<<<(unsigned)(((size_t)B_ * S_ * H_ * 64 + TPB - 1) / TPB), TPB, 0,
              stream>>>(Qb, H_);
  rope_bf16<<<(unsigned)(((size_t)B_ * S_ * HKV_ * 64 + TPB - 1) / TPB), TPB, 0,
              stream>>>(Kb, HKV_);
  transpose_v<<<(unsigned)((nKV + TPB - 1) / TPB), TPB, 0, stream>>>(Vb, VT);

  attn_fwd<<<B_ * H_ * (S_ / 32), 32, 0, stream>>>(Qb, Kb, VT, attn);

  // Output projection: (4096 x 4096) x (4096 x 4096) -> fp32
  dim3 go(D_ / 64, B_ * S_ / 64);
  gemm_bf16_wmma<float><<<go, 32, 0, stream>>>(attn, WoT, out, B_ * S_, D_, D_);
}
